// DensityGCNProcessor_22582938043088
// MI455X (gfx1250) — matrix operand order repaired and checked by reading, not compiled, not run on hardware
//
#include <hip/hip_runtime.h>

// ---------- types ----------
typedef __attribute__((ext_vector_type(8)))  __bf16 bf16x8;
typedef __attribute__((ext_vector_type(16))) __bf16 v16bf;
typedef __attribute__((ext_vector_type(8)))  float  v8f;

#define AS3 __attribute__((address_space(3)))

// ---------- problem constants (from reference setup_inputs) ----------
#define NB     8          // batches
#define HW2    2304       // 48*48 nodes per batch
#define NNODES (NB*HW2)   // 18432 total nodes
#define CIN    256
#define CHID   512
#define KNN    4

// GEMM tiling
#define TM     128        // block tile M
#define TN     128        // block tile N
#define ROWB   80         // staged row pitch in LDS (64B data + 16B pad)
#define TILEB  (128*ROWB) // one staged operand tile (10240 B)

// =====================================================================
// 1) kNN on densities: one block per 256 nodes of one batch.
//    Density row (2304 f32 = 9 KB) staged in LDS; each thread keeps a
//    sorted top-4 in registers. Strict '<' + ascending j reproduces the
//    stable-argsort tie-break. Also accumulates in-degree (push edges).
// =====================================================================
__global__ void knn_kernel(const float* __restrict__ dens,
                           int* __restrict__ nbr,
                           float* __restrict__ degf) {
    __shared__ float sdens[HW2];
    const int b     = blockIdx.x / 9;
    const int chunk = blockIdx.x % 9;
    const float* dbase = dens + b * HW2;
    for (int j = threadIdx.x; j < HW2; j += blockDim.x) sdens[j] = dbase[j];
    __syncthreads();

    const int i  = chunk * 256 + threadIdx.x;   // node within batch
    const float di = sdens[i];

    float d0 = 3.4e38f, d1 = 3.4e38f, d2 = 3.4e38f, d3 = 3.4e38f;
    int   i0 = 0, i1 = 0, i2 = 0, i3 = 0;
    for (int j = 0; j < HW2; ++j) {
        if (j == i) continue;
        float d = fabsf(di - sdens[j]);
        if (d < d3) {
            if (d < d0)      { d3=d2;i3=i2; d2=d1;i2=i1; d1=d0;i1=i0; d0=d;i0=j; }
            else if (d < d1) { d3=d2;i3=i2; d2=d1;i2=i1; d1=d;i1=j; }
            else if (d < d2) { d3=d2;i3=i2; d2=d;i2=j; }
            else             { d3=d;i3=j; }
        }
    }
    const int g  = b * HW2 + i;
    const int t0 = b * HW2 + i0, t1 = b * HW2 + i1;
    const int t2 = b * HW2 + i2, t3 = b * HW2 + i3;
    nbr[g*4+0] = t0; nbr[g*4+1] = t1; nbr[g*4+2] = t2; nbr[g*4+3] = t3;
    atomicAdd(&degf[t0], 1.0f); atomicAdd(&degf[t1], 1.0f);
    atomicAdd(&degf[t2], 1.0f); atomicAdd(&degf[t3], 1.0f);
}

// deg -> dis = rsqrt(deg + 1) (self loop adds 1; deg >= 1 always)
__global__ void dis_kernel(float* __restrict__ degf) {
    int i = blockIdx.x * blockDim.x + threadIdx.x;
    if (i < NNODES) degf[i] = rsqrtf(degf[i] + 1.0f);
}

// =====================================================================
// 2) Converts: x (NCHW f32 -> node-major bf16), W (K x N f32 -> N x K bf16)
// =====================================================================
__global__ void cvt_x_kernel(const float* __restrict__ feat, __bf16* __restrict__ xbf) {
    int idx  = blockIdx.x * blockDim.x + threadIdx.x;      // over NNODES*CIN
    int node = idx >> 8;
    int c    = idx & 255;
    int b    = node / HW2;
    int hw   = node % HW2;
    float v  = feat[((size_t)(b * CIN + c)) * HW2 + hw];
    xbf[idx] = (__bf16)v;
}

__global__ void cvt_w_kernel(const float* __restrict__ W, __bf16* __restrict__ Wt,
                             int K, int N) {
    int idx = blockIdx.x * blockDim.x + threadIdx.x;       // over K*N
    if (idx >= K * N) return;
    int n = idx / K;
    int k = idx % K;
    Wt[idx] = (__bf16)W[(size_t)k * N + n];
}

// =====================================================================
// 3) bf16 WMMA GEMM with async-to-LDS double buffering.
//    C[M,N] = A[M,K] * B[K,N], B given transposed [N,K].
//    256 threads = 8 waves; wave grid 4(M) x 2(N); wave tile 32x64
//    (2x4 fragments of v_wmma_f32_16x16x32_bf16). Block tile 128x128.
//    Per 32-K stage each thread async-copies 2x16B of A and 2x16B of B
//    (global_load_async_to_lds_b128, ASYNCcnt); fragments then come from
//    LDS via ds_load_b128. Stage k+1 DMA is issued before stage k math,
//    s_wait_asynccnt 0 + barrier close the stage -> DMA under the WMMAs.
//    Fragment lane layouts per CDNA5 ISA 7.12.2:
//      A: lane l<16 row=l, elems0-7 K=k0+h*8..+7, elems8-15 K=k0+16+h*8..
//      B: lane l<16 col=l, elems0-7 K=k0+h*8..+7, elems8-15 K=k0+16+h*8..   (h = lane>=16)
//      C: lane l: VGPR r -> (row = h*8+r, col = l&15)
// =====================================================================
__global__ void __launch_bounds__(256)
gemm_wmma_kernel(const __bf16* __restrict__ A,
                 const __bf16* __restrict__ Bt,
                 float* __restrict__ C,
                 int M, int N, int K) {
    // stage s: [ A tile | B tile ] at s*(2*TILEB); double buffered = 40 KB
    __shared__ __align__(16) char ldsbuf[2 * 2 * TILEB];
    AS3 char* lds = (AS3 char*)ldsbuf;

    const int tid  = threadIdx.x;
    const int lane = tid & 31;
    const int w    = tid >> 5;
    const int wm   = w & 3;
    const int wn   = w >> 2;
    const int half = lane >> 4;
    const int l16  = lane & 15;
    const int mBlock = blockIdx.y * TM;
    const int nBlock = blockIdx.x * TN;

    // copy assignment: granule = 16B (8 bf16); tile = 128 rows x 4 granules
    const int rowg = tid >> 2;        // rows rowg and rowg+64
    const int kcg  = (tid & 3) * 8;   // k element offset 0,8,16,24

    const int nk = K / 32;

    auto issue_stage = [&](int k0, int stage) {
        AS3 char* sb = lds + stage * (2 * TILEB);
        #pragma unroll
        for (int t = 0; t < 2; ++t) {
            const int row = rowg + t * 64;
            // A granule
            {
                AS3 char* dp = sb + row * ROWB + kcg * 2;
                const __bf16* gp = A + (size_t)(mBlock + row) * K + (k0 + kcg);
                unsigned la = (unsigned)(unsigned long long)dp;
                asm volatile("global_load_async_to_lds_b128 %0, %1, off"
                             :: "v"(la), "v"((unsigned long long)(size_t)gp)
                             : "memory");
            }
            // B granule
            {
                AS3 char* dp = sb + TILEB + row * ROWB + kcg * 2;
                const __bf16* gp = Bt + (size_t)(nBlock + row) * K + (k0 + kcg);
                unsigned lb = (unsigned)(unsigned long long)dp;
                asm volatile("global_load_async_to_lds_b128 %0, %1, off"
                             :: "v"(lb), "v"((unsigned long long)(size_t)gp)
                             : "memory");
            }
        }
    };

    const v8f zero = {0.f,0.f,0.f,0.f,0.f,0.f,0.f,0.f};
    v8f acc[2][4];
    #pragma unroll
    for (int mi = 0; mi < 2; ++mi)
        #pragma unroll
        for (int ni = 0; ni < 4; ++ni) acc[mi][ni] = zero;

    // prologue: stage 0 DMA
    issue_stage(0, 0);
    asm volatile("s_wait_asynccnt 0" ::: "memory");
    __syncthreads();

    for (int ks = 0; ks < nk; ++ks) {
        const int stage = ks & 1;
        if (ks + 1 < nk) issue_stage((ks + 1) * 32, stage ^ 1);

        AS3 const char* sA = lds + stage * (2 * TILEB);
        AS3 const char* sB = sA + TILEB;

        v16bf afrag[2], bfrag[4];
        #pragma unroll
        for (int mi = 0; mi < 2; ++mi) {
            const int row = wm * 32 + mi * 16 + l16;
            bf16x8 lo = *(AS3 const bf16x8*)(sA + row * ROWB + half * 16);
            bf16x8 hi = *(AS3 const bf16x8*)(sA + row * ROWB + 32 + half * 16);
            afrag[mi] = __builtin_shufflevector(lo, hi, 0,1,2,3,4,5,6,7,8,9,10,11,12,13,14,15);
        }
        #pragma unroll
        for (int ni = 0; ni < 4; ++ni) {
            const int col = wn * 64 + ni * 16 + l16;
            bf16x8 lo = *(AS3 const bf16x8*)(sB + col * ROWB + half * 16);
            bf16x8 hi = *(AS3 const bf16x8*)(sB + col * ROWB + 32 + half * 16);
            bfrag[ni] = __builtin_shufflevector(lo, hi, 0,1,2,3,4,5,6,7,8,9,10,11,12,13,14,15);
        }
        #pragma unroll
        for (int mi = 0; mi < 2; ++mi)
            #pragma unroll
            for (int ni = 0; ni < 4; ++ni)
                acc[mi][ni] = __builtin_amdgcn_wmma_f32_16x16x32_bf16(
                    false, afrag[mi], false, bfrag[ni],
                    (short)0, acc[mi][ni], false, false);

        if (ks + 1 < nk)
            asm volatile("s_wait_asynccnt 0" ::: "memory");
        __syncthreads();
    }

    #pragma unroll
    for (int mi = 0; mi < 2; ++mi)
        #pragma unroll
        for (int ni = 0; ni < 4; ++ni)
            #pragma unroll
            for (int r = 0; r < 8; ++r) {
                int row = mBlock + wm * 32 + mi * 16 + half * 8 + r;
                int col = nBlock + wn * 64 + ni * 16 + l16;
                C[(size_t)row * N + col] = acc[mi][ni][r];
            }
}

// =====================================================================
// 4) Edge aggregation: push h[s] into agg[t] for the 4 kNN targets.
//    One block per source node; channels across threads (coalesced
//    global_atomic_add_f32). Self loop handled in the finish pass.
// =====================================================================
__global__ void scatter_kernel(const float* __restrict__ h,
                               const int* __restrict__ nbr,
                               const float* __restrict__ dis,
                               float* __restrict__ agg, int C) {
    const int s  = blockIdx.x;
    const float ds = dis[s];
    const int t0 = nbr[s*4+0], t1 = nbr[s*4+1], t2 = nbr[s*4+2], t3 = nbr[s*4+3];
    const float n0 = ds*dis[t0], n1 = ds*dis[t1], n2 = ds*dis[t2], n3 = ds*dis[t3];
    const float* hs = h + (size_t)s * C;
    for (int c = threadIdx.x; c < C; c += blockDim.x) {
        float hv = hs[c];
        atomicAdd(&agg[(size_t)t0 * C + c], n0 * hv);
        atomicAdd(&agg[(size_t)t1 * C + c], n1 * hv);
        atomicAdd(&agg[(size_t)t2 * C + c], n2 * hv);
        atomicAdd(&agg[(size_t)t3 * C + c], n3 * hv);
    }
}

// layer-1 finish: + self loop + bias, ReLU, recast to bf16 for GEMM2
__global__ void finish1_kernel(const float* __restrict__ agg,
                               const float* __restrict__ h,
                               const float* __restrict__ dis,
                               const float* __restrict__ bias,
                               __bf16* __restrict__ ybf, int C) {
    int idx = blockIdx.x * blockDim.x + threadIdx.x;   // over NNODES*C
    int i = idx / C;
    int c = idx - i * C;
    float di = dis[i];
    float v  = agg[idx] + di * di * h[idx] + bias[c];
    ybf[idx] = (__bf16)fmaxf(v, 0.0f);
}

// layer-2 finish: + self loop + bias, ReLU, node-major -> NCHW into d_out
__global__ void finish2_kernel(const float* __restrict__ agg,
                               const float* __restrict__ h,
                               const float* __restrict__ dis,
                               const float* __restrict__ bias,
                               float* __restrict__ out) {
    int idx = blockIdx.x * blockDim.x + threadIdx.x;   // over NNODES*CIN
    int i = idx >> 8;
    int c = idx & 255;
    float di = dis[i];
    float v  = agg[idx] + di * di * h[idx] + bias[c];
    v = fmaxf(v, 0.0f);
    int b  = i / HW2;
    int hw = i % HW2;
    out[((size_t)(b * CIN + c)) * HW2 + hw] = v;
}

// =====================================================================
// launch
// =====================================================================
extern "C" void kernel_launch(void* const* d_in, const int* in_sizes, int n_in,
                              void* d_out, int out_size, void* d_ws, size_t ws_size,
                              hipStream_t stream) {
    const float* density = (const float*)d_in[0];   // [8,1,48,48]
    const float* feat    = (const float*)d_in[1];   // [8,256,48,48]
    const float* W1      = (const float*)d_in[2];   // [256,512]
    const float* b1      = (const float*)d_in[3];   // [512]
    const float* W2      = (const float*)d_in[4];   // [512,256]
    const float* b2      = (const float*)d_in[5];   // [256]
    float*       out     = (float*)d_out;           // [8,256,48,48]

    // workspace layout (all offsets 256B-aligned), ~100 MB total
    char* ws = (char*)d_ws;
    int*    nbr  = (int*)   (ws + 0);                       //  294912 B
    float*  disv = (float*) (ws + 294912);                  //   73728 B (deg -> dis in place)
    __bf16* xbf  = (__bf16*)(ws + 368640);                  // 9437184 B
    __bf16* w1t  = (__bf16*)(ws + 9805824);                 //  262144 B
    __bf16* w2t  = (__bf16*)(ws + 10067968);                //  262144 B
    float*  h    = (float*) (ws + 10330112);                // 37748736 B (h1, reused for h2)
    float*  agg  = (float*) (ws + 48078848);                // 37748736 B (agg1, reused for agg2)
    __bf16* ybf  = (__bf16*)(ws + 85827584);                // 18874368 B

    // ---- graph / edges ----
    hipMemsetAsync(disv, 0, (size_t)NNODES * 4, stream);
    knn_kernel<<<NB * 9, 256, 0, stream>>>(density, nbr, disv);
    dis_kernel<<<(NNODES + 255) / 256, 256, 0, stream>>>(disv);

    // ---- bf16 staging ----
    cvt_x_kernel<<<(NNODES * CIN) / 256, 256, 0, stream>>>(feat, xbf);
    cvt_w_kernel<<<(CIN * CHID + 255) / 256, 256, 0, stream>>>(W1, w1t, CIN, CHID);
    cvt_w_kernel<<<(CHID * CIN + 255) / 256, 256, 0, stream>>>(W2, w2t, CHID, CIN);

    // ---- layer 1: GEMM + aggregate + bias/ReLU ----
    gemm_wmma_kernel<<<dim3(CHID / TN, NNODES / TM), 256, 0, stream>>>(
        xbf, w1t, h, NNODES, CHID, CIN);
    hipMemsetAsync(agg, 0, (size_t)NNODES * CHID * 4, stream);
    scatter_kernel<<<NNODES, 256, 0, stream>>>(h, nbr, disv, agg, CHID);
    finish1_kernel<<<(NNODES * CHID) / 256, 256, 0, stream>>>(agg, h, disv, b1, ybf, CHID);

    // ---- layer 2: GEMM + aggregate + bias/ReLU + permute to NCHW ----
    gemm_wmma_kernel<<<dim3(CIN / TN, NNODES / TM), 256, 0, stream>>>(
        ybf, w2t, h, NNODES, CIN, CHID);
    hipMemsetAsync(agg, 0, (size_t)NNODES * CIN * 4, stream);
    scatter_kernel<<<NNODES, 256, 0, stream>>>(h, nbr, disv, agg, CIN);
    finish2_kernel<<<(NNODES * CIN) / 256, 256, 0, stream>>>(agg, h, disv, b2, out);
}
